// PointTransformerDownBlock_5617817224089
// MI455X (gfx1250) — matrix operand order repaired and checked by reading, not compiled, not run on hardware
//
#include <hip/hip_runtime.h>

typedef __attribute__((ext_vector_type(16))) _Float16 v16h;
typedef __attribute__((ext_vector_type(8)))  _Float16 v8h;
typedef __attribute__((ext_vector_type(8)))  float    v8f;

#define CB  4      // batch
#define CN  8192   // input points
#define CS  2048   // sampled points
#define CK  16     // neighbors
#define CDIN 64
#define CDOUT 128
#define CMID 32

// ---------------------------------------------------------------------------
// Farthest point sampling: one block per batch, dist array in LDS (32 KB).
// Matches jax reference: idx[0]=0, argmax picks first (lowest index) max.
// ---------------------------------------------------------------------------
#define FPS_T 256
__global__ void k_fps(const float* __restrict__ xyz, int* __restrict__ fps_idx) {
  __shared__ float dist[CN];
  __shared__ float rmax[FPS_T];
  __shared__ int   rarg[FPS_T];
  __shared__ int   s_far;
  const int b = blockIdx.x;
  const int t = threadIdx.x;
  const float* px = xyz + (size_t)b * 3 * CN;
  const float* py = px + CN;
  const float* pz = py + CN;
  for (int n = t; n < CN; n += FPS_T) dist[n] = 1e10f;
  if (t == 0) s_far = 0;
  __syncthreads();
  for (int s = 0; s < CS; ++s) {
    const int far = s_far;
    if (t == 0) fps_idx[b * CS + s] = far;
    const float cx = px[far], cy = py[far], cz = pz[far];
    float best = -1.0f; int bidx = 0;
    for (int n = t; n < CN; n += FPS_T) {
      float dx = px[n] - cx, dy = py[n] - cy, dz = pz[n] - cz;
      float d  = dx * dx + dy * dy + dz * dz;
      float dn = fminf(dist[n], d);
      dist[n] = dn;
      if (dn > best) { best = dn; bidx = n; }   // strict > keeps lowest index
    }
    rmax[t] = best; rarg[t] = bidx;
    __syncthreads();
    for (int w = FPS_T / 2; w > 0; w >>= 1) {
      if (t < w) {
        float ov = rmax[t + w]; int oi = rarg[t + w];
        if (ov > rmax[t] || (ov == rmax[t] && oi < rarg[t])) { rmax[t] = ov; rarg[t] = oi; }
      }
      __syncthreads();
    }
    if (t == 0) s_far = rarg[0];
    __syncthreads();
  }
}

// gather new_xyz [B,3,S] straight into d_out
__global__ void k_gather_newxyz(const float* __restrict__ xyz, const int* __restrict__ fps_idx,
                                float* __restrict__ nxyz) {
  int t = blockIdx.x * blockDim.x + threadIdx.x;  // B*S
  int s = t % CS, b = t / CS;
  int g = fps_idx[t];
  for (int c = 0; c < 3; ++c)
    nxyz[((size_t)(b * 3 + c)) * CS + s] = xyz[((size_t)(b * 3 + c)) * CN + g];
}

// ---------------------------------------------------------------------------
// KNN: one thread per query, top-16 via fully unrolled register insertion.
// Strict < keeps lower index on ties (matches lax.top_k stability).
// ---------------------------------------------------------------------------
__global__ void k_knn(const float* __restrict__ Q, const float* __restrict__ P,
                      int* __restrict__ idx, int Npts) {
  int t = blockIdx.x * blockDim.x + threadIdx.x;  // B*S
  int s = t % CS, b = t / CS;
  const float* qb = Q + (size_t)b * 3 * CS;
  float qx = qb[s], qy = qb[CS + s], qz = qb[2 * CS + s];
  const float* pb = P + (size_t)b * 3 * Npts;
  float bd[CK]; int bi[CK];
#pragma unroll
  for (int j = 0; j < CK; ++j) { bd[j] = 3.4e38f; bi[j] = 0; }
  for (int n = 0; n < Npts; ++n) {
    float dx = qx - pb[n], dy = qy - pb[Npts + n], dz = qz - pb[2 * Npts + n];
    float d = dx * dx + dy * dy + dz * dz;
    if (d < bd[CK - 1]) {
      float dj = d; int nj = n;
#pragma unroll
      for (int j = 0; j < CK; ++j) {
        if (dj < bd[j]) { float td = bd[j]; int ti = bi[j]; bd[j] = dj; bi[j] = nj; dj = td; nj = ti; }
      }
    }
  }
#pragma unroll
  for (int j = 0; j < CK; ++j) idx[(size_t)t * CK + j] = bi[j];
}

// ---------------------------------------------------------------------------
// Build grouped TD input X [ncol=B*S*K][Kpad=96] f16 (points gather + rel xyz, zero pad)
// ---------------------------------------------------------------------------
__global__ void k_build_td_x(const float* __restrict__ points, const float* __restrict__ xyz,
                             const float* __restrict__ nxyz, const int* __restrict__ gidx,
                             _Float16* __restrict__ Xg) {
  int col = blockIdx.x * blockDim.x + threadIdx.x;  // B*S*K
  int bs = col >> 4;
  int s = bs % CS, b = bs / CS;
  int g = gidx[col];
  _Float16* out = Xg + (size_t)col * 96;
  const float* pb = points + (size_t)b * CDIN * CN;
  for (int c = 0; c < CDIN; ++c) out[c] = (_Float16)pb[(size_t)c * CN + g];
  const float* xb = xyz + (size_t)b * 3 * CN;
  const float* nb = nxyz + (size_t)b * 3 * CS;
#pragma unroll
  for (int c = 0; c < 3; ++c)
    out[CDIN + c] = (_Float16)(xb[(size_t)c * CN + g] - nb[(size_t)c * CS + s]);
  for (int c = CDIN + 3; c < 96; ++c) out[c] = (_Float16)0.0f;
}

// weight f32 [M][Kin] row-major -> f16 [M][Kpad] zero-padded
__global__ void k_w2h(const float* __restrict__ src, _Float16* __restrict__ dst,
                      int M, int Kin, int Kpad) {
  int t = blockIdx.x * blockDim.x + threadIdx.x;
  if (t >= M * Kpad) return;
  int m = t / Kpad, k = t % Kpad;
  dst[t] = (k < Kin) ? (_Float16)src[m * Kin + k] : (_Float16)0.0f;
}

// ---------------------------------------------------------------------------
// Single-wave WMMA GEMM, MT 16x16 m-tiles per wave (B fragment reused MT times,
// cutting activation re-reads by MT; weights stay L0-resident).
//   A: weights f16 row-major [M][ldk]   (ldk multiple of 8)
//   B: activations f16 col-major [ncol][ldk] (ldk multiple of 16)
//   Cf: optional f32 row-major [M][ncol];  Ch: optional f16 col-major [ncol][M]
// Fragment k order: contiguous k = k0 + 16*hi + e in both A and B (wave32).
// ---------------------------------------------------------------------------
template <int MT>
__global__ void k_gemm(const _Float16* __restrict__ A, const _Float16* __restrict__ B,
                       float* __restrict__ Cf, _Float16* __restrict__ Ch,
                       const float* __restrict__ bias,
                       int M, int ncol, int ldk, int relu) {
  const int lane = threadIdx.x;        // block = 32 threads = 1 wave
  const int lo = lane & 15;
  const int hi = lane >> 4;
  const int nBase = blockIdx.x * 16;
  const int mBase0 = blockIdx.y * MT * 16;
  const _Float16* bcol = B + (size_t)(nBase + lo) * ldk;
  v8f zero = {};
  v8f acc[MT];
#pragma unroll
  for (int j = 0; j < MT; ++j) acc[j] = zero;
  for (int k0 = 0; k0 < ldk; k0 += 32) {
    v16h bf = *(const v16h*)(bcol + k0 + hi * 16);     // N=lo, K=k0+hi*16+e
#pragma unroll
    for (int j = 0; j < MT; ++j) {
      const _Float16* arow = A + (size_t)(mBase0 + j * 16 + lo) * ldk + k0;
      v8h a0 = *(const v8h*)(arow + hi * 16);          // lanes0-15: K+0..7 ; lanes16-31: K+16..23
      v8h a1 = *(const v8h*)(arow + hi * 16 + 8);      // lanes0-15: K+8..15; lanes16-31: K+24..31
      v16h a;
#pragma unroll
      for (int i = 0; i < 8; ++i) { a[i] = a0[i]; a[8 + i] = a1[i]; }
      acc[j] = __builtin_amdgcn_wmma_f32_16x16x32_f16(false, a, false, bf,
                                                      (short)0, acc[j], false, false);
    }
  }
  if (Cf) {
#pragma unroll
    for (int j = 0; j < MT; ++j) {
#pragma unroll
      for (int r = 0; r < 8; ++r) {
        int m = mBase0 + j * 16 + hi * 8 + r;
        float v = acc[j][r];
        if (bias) v += bias[m];
        if (relu) v = fmaxf(v, 0.0f);
        Cf[(size_t)m * ncol + nBase + lo] = v;
      }
    }
  }
  if (Ch) {
#pragma unroll
    for (int j = 0; j < MT; ++j) {
      v8h h;
#pragma unroll
      for (int r = 0; r < 8; ++r) {
        int m = mBase0 + j * 16 + hi * 8 + r;
        float v = acc[j][r];
        if (bias) v += bias[m];
        if (relu) v = fmaxf(v, 0.0f);
        h[r] = (_Float16)v;
      }
      *(v8h*)(Ch + (size_t)(nBase + lo) * M + mBase0 + j * 16 + hi * 8) = h;
    }
  }
}

// ---------------------------------------------------------------------------
// BatchNorm stats (training mode, over B*S*K per channel) — deterministic 2-stage
// ---------------------------------------------------------------------------
__global__ void k_bn_part(const float* __restrict__ Y, float* __restrict__ part, int ncol, int nchunk) {
  __shared__ float ssum[256], ssq[256];
  int c = blockIdx.x, chunk = blockIdx.y, t = threadIdx.x;
  int per = ncol / nchunk;
  const float* row = Y + (size_t)c * ncol + (size_t)chunk * per;
  float s0 = 0.0f, s1 = 0.0f;
  for (int i = t; i < per; i += 256) { float v = row[i]; s0 += v; s1 += v * v; }
  ssum[t] = s0; ssq[t] = s1;
  __syncthreads();
  for (int w = 128; w > 0; w >>= 1) {
    if (t < w) { ssum[t] += ssum[t + w]; ssq[t] += ssq[t + w]; }
    __syncthreads();
  }
  if (t == 0) {
    part[((size_t)c * nchunk + chunk) * 2]     = ssum[0];
    part[((size_t)c * nchunk + chunk) * 2 + 1] = ssq[0];
  }
}

__global__ void k_bn_final(const float* __restrict__ part, float* __restrict__ stats,
                           int nchunk, int ncol) {
  __shared__ float a[64], q[64];
  int c = blockIdx.x, t = threadIdx.x;   // block = 64
  a[t] = (t < nchunk) ? part[((size_t)c * nchunk + t) * 2] : 0.0f;
  q[t] = (t < nchunk) ? part[((size_t)c * nchunk + t) * 2 + 1] : 0.0f;
  __syncthreads();
  for (int w = 32; w > 0; w >>= 1) {
    if (t < w) { a[t] += a[t + w]; q[t] += q[t + w]; }
    __syncthreads();
  }
  if (t == 0) {
    float mean = a[0] / (float)ncol;
    float var  = q[0] / (float)ncol - mean * mean;
    stats[c * 2] = mean; stats[c * 2 + 1] = var;
  }
}

// BN + ReLU + max over K  -> pts [B,128,S] (written into d_out)
__global__ void k_td_finish(const float* __restrict__ Y, const float* __restrict__ stats,
                            const float* __restrict__ gamma, const float* __restrict__ beta,
                            float* __restrict__ pts, int ncol) {
  int t = blockIdx.x * blockDim.x + threadIdx.x;   // B*128*S, t = (b*128+o)*S+s
  int s = t % CS; int rest = t / CS; int o = rest % CDOUT; int b = rest / CDOUT;
  float mean = stats[o * 2], var = stats[o * 2 + 1];
  float inv = rsqrtf(var + 1e-5f);
  float g = gamma[o], be = beta[o];
  const float* yp = Y + (size_t)o * ncol + ((size_t)(b * CS + s)) * CK;
  float mx = -3.4e38f;
#pragma unroll
  for (int k = 0; k < CK; ++k) {
    float v = g * (yp[k] - mean) * inv + be;
    v = fmaxf(v, 0.0f);
    mx = fmaxf(mx, v);
  }
  pts[t] = mx;
}

// ---------------------------------------------------------------------------
// Positional MLP: rel[col][32] = w2·relu(w1·(center - neighbor) + b1) + b2
// ---------------------------------------------------------------------------
__global__ void k_pos_rel(const float* __restrict__ nxyz, const int* __restrict__ sgidx,
                          const float* __restrict__ w1, const float* __restrict__ b1,
                          const float* __restrict__ w2, const float* __restrict__ b2,
                          float* __restrict__ rel) {
  int col = blockIdx.x * blockDim.x + threadIdx.x;  // B*S*K
  int bs = col >> 4;
  int s = bs % CS, b = bs / CS;
  int g = sgidx[col];
  const float* nb = nxyz + (size_t)b * 3 * CS;
  float dx = nb[s] - nb[g];
  float dy = nb[CS + s] - nb[CS + g];
  float dz = nb[2 * CS + s] - nb[2 * CS + g];
  float h[64];
#pragma unroll
  for (int j = 0; j < 64; ++j) {
    float v = w1[j * 3] * dx + w1[j * 3 + 1] * dy + w1[j * 3 + 2] * dz + b1[j];
    h[j] = fmaxf(v, 0.0f);
  }
  float* out = rel + (size_t)col * CMID;
  for (int c = 0; c < CMID; ++c) {
    float acc = b2[c];
#pragma unroll
    for (int j = 0; j < 64; ++j) acc += w2[c * 64 + j] * h[j];
    out[c] = acc;
  }
}

// pts f32 [B,128,S] -> f16 col-major [B*S][128]
__global__ void k_pts_to_f16(const float* __restrict__ pts, _Float16* __restrict__ out) {
  int t = blockIdx.x * blockDim.x + threadIdx.x;  // B*S*128
  int c = t % CDOUT; int bs = t / CDOUT; int s = bs % CS; int b = bs / CS;
  out[t] = (_Float16)pts[((size_t)(b * CDOUT + c)) * CS + s];
}

// X1[col][32] = f16(q[c,s] - k[c,neighbor] + rel[col][c])
__global__ void k_build_x1(const float* __restrict__ qkv, const int* __restrict__ sgidx,
                           const float* __restrict__ rel, _Float16* __restrict__ X1, int ncol2) {
  int t = blockIdx.x * blockDim.x + threadIdx.x;  // B*S*K*32
  int c = t % CMID; int col = t / CMID;
  int bs = col >> 4;
  int g = sgidx[col];
  int b = bs / CS;
  float qv = qkv[(size_t)c * ncol2 + bs];
  float kv = qkv[(size_t)(CMID + c) * ncol2 + b * CS + g];
  X1[t] = (_Float16)(qv - kv + rel[t]);
}

// softmax over K + weighted sum of (v_gathered + rel) -> agg f16 [B*S][32]
__global__ void k_softmax_agg(const float* __restrict__ sim, const float* __restrict__ qkv,
                              const int* __restrict__ sgidx, const float* __restrict__ rel,
                              _Float16* __restrict__ agg, int ncol1, int ncol2) {
  int t = blockIdx.x * blockDim.x + threadIdx.x;  // B*S*32
  int c = t % CMID; int bs = t / CMID; int b = bs / CS;
  int col0 = bs * CK;
  const float* sp = sim + (size_t)c * ncol1 + col0;
  float sv[CK];
  float m = -3.4e38f;
#pragma unroll
  for (int k = 0; k < CK; ++k) { sv[k] = sp[k]; m = fmaxf(m, sv[k]); }
  float sum = 0.0f;
#pragma unroll
  for (int k = 0; k < CK; ++k) { sv[k] = __expf(sv[k] - m); sum += sv[k]; }
  float inv = 1.0f / sum;
  const float* vp = qkv + (size_t)(2 * CMID + c) * ncol2 + (size_t)b * CS;
  float acc = 0.0f;
#pragma unroll
  for (int k = 0; k < CK; ++k) {
    int g = sgidx[col0 + k];
    float v = vp[g] + rel[(size_t)(col0 + k) * CMID + c];
    acc += sv[k] * inv * v;
  }
  agg[t] = (_Float16)acc;
}

// pts += delta  (delta f32 row-major [128][B*S])
__global__ void k_update_pts(float* __restrict__ pts, const float* __restrict__ delta, int ncol2) {
  int t = blockIdx.x * blockDim.x + threadIdx.x;  // t = (b*128+o)*S+s
  int s = t % CS; int rest = t / CS; int o = rest % CDOUT; int b = rest / CDOUT;
  pts[t] += delta[(size_t)o * ncol2 + b * CS + s];
}

// ---------------------------------------------------------------------------
extern "C" void kernel_launch(void* const* d_in, const int* in_sizes, int n_in,
                              void* d_out, int out_size, void* d_ws, size_t ws_size,
                              hipStream_t stream) {
  (void)n_in; (void)out_size; (void)ws_size;
  const int NCOL1 = CB * CS * CK;   // 131072
  const int NCOL2 = CB * CS;        // 8192
  const int A = in_sizes[6] / (CMID * CDOUT);

  const float* xyz        = (const float*)d_in[0];
  const float* points     = (const float*)d_in[1];
  const float* td_w       = (const float*)d_in[2];
  const float* td_b       = (const float*)d_in[3];
  const float* td_gamma   = (const float*)d_in[4];
  const float* td_beta    = (const float*)d_in[5];
  const float* bw_before  = (const float*)d_in[6];
  const float* bb_before  = (const float*)d_in[7];
  const float* bw_qkv     = (const float*)d_in[8];
  const float* bw_pos1    = (const float*)d_in[9];
  const float* bb_pos1    = (const float*)d_in[10];
  const float* bw_pos2    = (const float*)d_in[11];
  const float* bb_pos2    = (const float*)d_in[12];
  const float* bw_attn1   = (const float*)d_in[13];
  const float* bb_attn1   = (const float*)d_in[14];
  const float* bw_attn2   = (const float*)d_in[15];
  const float* bb_attn2   = (const float*)d_in[16];
  const float* bw_after   = (const float*)d_in[17];
  const float* bb_after   = (const float*)d_in[18];

  float* out_nxyz = (float*)d_out;                   // [B,3,S]
  float* out_pts  = out_nxyz + (size_t)CB * 3 * CS;  // [B,128,S]

  // ---- workspace carving (256-byte aligned) ----
  size_t off = 0;
  auto carve = [&](size_t bytes) -> void* {
    off = (off + 255) & ~(size_t)255;
    void* p = (char*)d_ws + off;
    off += bytes;
    return p;
  };
  int*       fps_idx = (int*)carve((size_t)CB * CS * 4);
  int*       gidx    = (int*)carve((size_t)NCOL1 * 4);
  int*       sgidx   = (int*)carve((size_t)NCOL1 * 4);
  _Float16*  wtd     = (_Float16*)carve((size_t)CDOUT * 96 * 2);
  _Float16*  wbefore = (_Float16*)carve((size_t)A * CMID * CDOUT * 2);
  _Float16*  wqkv    = (_Float16*)carve((size_t)A * 3 * CMID * CMID * 2);
  _Float16*  wattn1  = (_Float16*)carve((size_t)A * 4 * CMID * CMID * 2);
  _Float16*  wattn2  = (_Float16*)carve((size_t)A * CMID * 4 * CMID * 2);
  _Float16*  wafter  = (_Float16*)carve((size_t)A * CDOUT * CMID * 2);
  float*     stats   = (float*)carve((size_t)CDOUT * 2 * 4);
  float*     bnpart  = (float*)carve((size_t)CDOUT * 64 * 2 * 4);
  _Float16*  pts16   = (_Float16*)carve((size_t)NCOL2 * CDOUT * 2);
  _Float16*  pf16    = (_Float16*)carve((size_t)NCOL2 * CMID * 2);
  float*     qkvf    = (float*)carve((size_t)3 * CMID * NCOL2 * 4);
  _Float16*  aggf    = (_Float16*)carve((size_t)NCOL2 * CMID * 2);
  float*     delta   = (float*)carve((size_t)CDOUT * NCOL2 * 4);

  // overlapped arena: TD stage (Xg 25.2MB + Y 67.1MB) vs attention stage
  // (rel 16.8MB + X1 8.4MB + H1f16 33.6MB + SIM 16.8MB) — stages are sequential.
  off = (off + 255) & ~(size_t)255;
  char* arena = (char*)d_ws + off;
  auto align256 = [](size_t v) { return (v + 255) & ~(size_t)255; };
  _Float16* Xg  = (_Float16*)arena;
  float*    Y   = (float*)(arena + align256((size_t)NCOL1 * 96 * 2));
  float*    rel = (float*)arena;
  _Float16* X1  = (_Float16*)((char*)rel + align256((size_t)NCOL1 * CMID * 4));
  _Float16* H1  = (_Float16*)((char*)X1 + align256((size_t)NCOL1 * CMID * 2));
  float*    SIM = (float*)((char*)H1 + align256((size_t)NCOL1 * (4 * CMID) * 2));

  // ---- stage 1: FPS + gathers + KNNs ----
  k_fps<<<CB, FPS_T, 0, stream>>>(xyz, fps_idx);
  k_gather_newxyz<<<(CB * CS) / 256, 256, 0, stream>>>(xyz, fps_idx, out_nxyz);
  k_knn<<<(CB * CS) / 256, 256, 0, stream>>>(out_nxyz, xyz, gidx, CN);
  k_knn<<<(CB * CS) / 256, 256, 0, stream>>>(out_nxyz, out_nxyz, sgidx, CS);

  // ---- weight conversion to padded f16 ----
  k_w2h<<<(CDOUT * 96 + 255) / 256, 256, 0, stream>>>(td_w, wtd, CDOUT, CDIN + 3, 96);
  for (int i = 0; i < A; ++i) {
    k_w2h<<<(CMID * CDOUT + 255) / 256, 256, 0, stream>>>(
        bw_before + (size_t)i * CMID * CDOUT, wbefore + (size_t)i * CMID * CDOUT, CMID, CDOUT, CDOUT);
    k_w2h<<<(3 * CMID * CMID + 255) / 256, 256, 0, stream>>>(
        bw_qkv + (size_t)i * 3 * CMID * CMID, wqkv + (size_t)i * 3 * CMID * CMID, 3 * CMID, CMID, CMID);
    k_w2h<<<(4 * CMID * CMID + 255) / 256, 256, 0, stream>>>(
        bw_attn1 + (size_t)i * 4 * CMID * CMID, wattn1 + (size_t)i * 4 * CMID * CMID, 4 * CMID, CMID, CMID);
    k_w2h<<<(CMID * 4 * CMID + 255) / 256, 256, 0, stream>>>(
        bw_attn2 + (size_t)i * CMID * 4 * CMID, wattn2 + (size_t)i * CMID * 4 * CMID, CMID, 4 * CMID, 4 * CMID);
    k_w2h<<<(CDOUT * CMID + 255) / 256, 256, 0, stream>>>(
        bw_after + (size_t)i * CDOUT * CMID, wafter + (size_t)i * CDOUT * CMID, CDOUT, CMID, CMID);
  }

  // ---- stage 2: transition-down conv (WMMA) + BN + ReLU + maxpool ----
  k_build_td_x<<<NCOL1 / 256, 256, 0, stream>>>(points, xyz, out_nxyz, gidx, Xg);
  k_gemm<8><<<dim3(NCOL1 / 16, 1), 32, 0, stream>>>(
      wtd, Xg, Y, nullptr, td_b, CDOUT, NCOL1, 96, 0);
  k_bn_part<<<dim3(CDOUT, 64), 256, 0, stream>>>(Y, bnpart, NCOL1, 64);
  k_bn_final<<<CDOUT, 64, 0, stream>>>(bnpart, stats, 64, NCOL1);
  k_td_finish<<<(CB * CDOUT * CS) / 256, 256, 0, stream>>>(Y, stats, td_gamma, td_beta, out_pts, NCOL1);

  // ---- stage 3: attention res-blocks ----
  for (int i = 0; i < A; ++i) {
    k_pos_rel<<<NCOL1 / 256, 256, 0, stream>>>(
        out_nxyz, sgidx,
        bw_pos1 + (size_t)i * 64 * 3, bb_pos1 + (size_t)i * 64,
        bw_pos2 + (size_t)i * CMID * 64, bb_pos2 + (size_t)i * CMID, rel);
    k_pts_to_f16<<<(NCOL2 * CDOUT) / 256, 256, 0, stream>>>(out_pts, pts16);
    // p = before_w @ pts + b  -> f16 col-major [ncol2][32]
    k_gemm<2><<<dim3(NCOL2 / 16, 1), 32, 0, stream>>>(
        wbefore + (size_t)i * CMID * CDOUT, pts16, nullptr, pf16,
        bb_before + (size_t)i * CMID, CMID, NCOL2, CDOUT, 0);
    // qkv = qkv_w @ p  (no bias) -> f32 row-major [96][ncol2]
    k_gemm<2><<<dim3(NCOL2 / 16, 3), 32, 0, stream>>>(
        wqkv + (size_t)i * 3 * CMID * CMID, pf16, qkvf, nullptr,
        nullptr, 3 * CMID, NCOL2, CMID, 0);
    k_build_x1<<<((size_t)NCOL1 * CMID) / 256, 256, 0, stream>>>(qkvf, sgidx, rel, X1, NCOL2);
    // h1 = relu(attn_w1 @ (qk_rel+rel) + b1) -> f16 col-major [ncol1][128]
    k_gemm<8><<<dim3(NCOL1 / 16, 1), 32, 0, stream>>>(
        wattn1 + (size_t)i * 4 * CMID * CMID, X1, nullptr, H1,
        bb_attn1 + (size_t)i * 4 * CMID, 4 * CMID, NCOL1, CMID, 1);
    // sim = attn_w2 @ h1 + b2 -> f32 row-major [32][ncol1]
    k_gemm<2><<<dim3(NCOL1 / 16, 1), 32, 0, stream>>>(
        wattn2 + (size_t)i * CMID * 4 * CMID, H1, SIM, nullptr,
        bb_attn2 + (size_t)i * CMID, CMID, NCOL1, 4 * CMID, 0);
    k_softmax_agg<<<(NCOL2 * CMID) / 256, 256, 0, stream>>>(SIM, qkvf, sgidx, rel, aggf, NCOL1, NCOL2);
    // delta = after_w @ agg + after_b -> f32 row-major [128][ncol2]
    k_gemm<8><<<dim3(NCOL2 / 16, 1), 32, 0, stream>>>(
        wafter + (size_t)i * CDOUT * CMID, aggf, delta, nullptr,
        bb_after + (size_t)i * CDOUT, CDOUT, NCOL2, CMID, 0);
    k_update_pts<<<(CB * CDOUT * CS) / 256, 256, 0, stream>>>(out_pts, delta, NCOL2);
  }
}
